// VisionTransformer_35347580846339
// MI455X (gfx1250) — matrix operand order repaired and checked by reading, not compiled, not run on hardware
//
#include <hip/hip_runtime.h>
#include <hip/hip_bf16.h>
#include <math.h>

// ---------------- model constants ----------------
#define BC   16
#define DC   384
#define HC   6
#define HDC  64
#define HIDC 1536

typedef __attribute__((ext_vector_type(16))) __bf16 v16bf;
typedef __attribute__((ext_vector_type(8)))  float  v8f;

// packed fp32->bf16 (RNE) using the HW converter: 1 VALU op per 2 elements
__device__ __forceinline__ unsigned int pack2bf(float x, float y) {
  unsigned int r;
  asm("v_cvt_pk_bf16_f32 %0, %1, %2" : "=v"(r) : "v"(x), "v"(y));
  return r;
}

// Fragment-order index maps (ISA 7.12.2, 16-bit operands, wave32).
// A 16x32 fragment (512 u16): element (r, k)
__device__ __forceinline__ int afrag_idx(int r, int k) {
  int lane = r + (((k & 15) >= 8) ? 16 : 0);
  int e = (k < 16) ? (k & 7) : 8 + (k & 7);
  return lane * 16 + e;
}
// B 32x16 fragment (512 u16): element (k, n)
__device__ __forceinline__ int bfrag_idx(int k, int n) {
  int lane = n + (((k & 15) >= 8) ? 16 : 0);
  int e = (k < 16) ? (k & 7) : 8 + (k & 7);
  return lane * 16 + e;
}
__device__ __forceinline__ v16bf ldfrag(const unsigned short* p) {
  return *(const v16bf*)p;                   // 32B -> 2x ds_load_b128
}

// ---------------------------------------------------------------
// Generic GEMM: C[M,N] = act(A[M,K] @ W + bias) + resid
//   W row-major (K,N) if wt==0, else (N,K) (transposed access)
//   act: 0 = none, 1 = exact GELU
// 256 threads = 8 waves; block tile 128x64; wave tile 32x32 (2x2 WMMA)
// LDS tiles in WMMA fragment order; staging: b128 load + 2 cvt + b64 store
// ---------------------------------------------------------------
__global__ __launch_bounds__(256)
void gemm_bf16_wmma(const float* __restrict__ A, const float* __restrict__ W,
                    const float* __restrict__ bias, const float* __restrict__ resid,
                    float* __restrict__ C, int M, int N, int K, int wt, int act) {
  __shared__ __align__(32) unsigned short As[8 * 512];   // 8 row-subtiles of 16x32
  __shared__ __align__(32) unsigned short Bs[4 * 512];   // 4 col-subtiles of 32x16

  const int tid  = threadIdx.x;
  const int lane = tid & 31;
  const int wid  = tid >> 5;
  const int wm   = wid >> 1;               // 0..3
  const int wn   = wid & 1;                // 0..1
  const int m_blk = blockIdx.y * 128;
  const int n_blk = blockIdx.x * 64;

  v8f acc[2][2] = {};

  for (int k0 = 0; k0 < K; k0 += 32) {
    // ---- stage A tile: 1024 float4 quads ----
    #pragma unroll
    for (int i = 0; i < 4; ++i) {
      int p = tid + i * 256;               // 0..1023
      int r = p >> 3;                      // 0..127
      int cq = (p & 7) * 4;                // 0,4,..,28
      int gr = m_blk + r;
      int ok = gr < M;
      const float4 v4 = *(const float4*)(A + (long long)(ok ? gr : 0) * K + k0 + cq);
      unsigned int px = pack2bf(v4.x, v4.y);   // unconditional asm
      unsigned int py = pack2bf(v4.z, v4.w);
      uint2 pk;
      pk.x = ok ? px : 0u;                     // cndmask, no exec branch
      pk.y = ok ? py : 0u;
      int idx = (r >> 4) * 512 + afrag_idx(r & 15, cq);
      *(uint2*)&As[idx] = pk;
    }
    // ---- stage B tile: 512 quads along k ----
    #pragma unroll
    for (int i = 0; i < 2; ++i) {
      int p = tid + i * 256;               // 0..511
      int kq = (p >> 6) * 4;               // 0,4,..,28
      int n  = p & 63;
      int gn = n_blk + n;
      int ok = gn < N;
      float x0, x1, x2, x3;
      if (wt) {
        const float4 v4 = *(const float4*)(W + (long long)(ok ? gn : 0) * K + k0 + kq);
        x0 = v4.x; x1 = v4.y; x2 = v4.z; x3 = v4.w;
      } else {
        const float* base = W + (long long)(k0 + kq) * N + (ok ? gn : 0);
        x0 = base[0]; x1 = base[N]; x2 = base[2 * N]; x3 = base[3 * N];
      }
      unsigned int px = pack2bf(x0, x1);
      unsigned int py = pack2bf(x2, x3);
      uint2 pk;
      pk.x = ok ? px : 0u;
      pk.y = ok ? py : 0u;
      int idx = (n >> 4) * 512 + bfrag_idx(kq, n & 15);
      *(uint2*)&Bs[idx] = pk;
    }
    __syncthreads();

    const unsigned short* Ab = As + (wm * 2) * 512 + lane * 16;
    const unsigned short* Bb = Bs + (wn * 2) * 512 + lane * 16;
    v16bf a0 = ldfrag(Ab);
    v16bf a1 = ldfrag(Ab + 512);
    v16bf b0 = ldfrag(Bb);
    v16bf b1 = ldfrag(Bb + 512);
    acc[0][0] = __builtin_amdgcn_wmma_f32_16x16x32_bf16(false, a0, false, b0, (short)0, acc[0][0], false, false);
    acc[0][1] = __builtin_amdgcn_wmma_f32_16x16x32_bf16(false, a0, false, b1, (short)0, acc[0][1], false, false);
    acc[1][0] = __builtin_amdgcn_wmma_f32_16x16x32_bf16(false, a1, false, b0, (short)0, acc[1][0], false, false);
    acc[1][1] = __builtin_amdgcn_wmma_f32_16x16x32_bf16(false, a1, false, b1, (short)0, acc[1][1], false, false);
    __syncthreads();
  }

  // epilogue: C/D layout -> rows (lane>>4)*8 + r, col lane&15
  const int rb = (lane >> 4) * 8;
  const int cb = lane & 15;
  #pragma unroll
  for (int ti = 0; ti < 2; ++ti)
    #pragma unroll
    for (int tj = 0; tj < 2; ++tj)
      #pragma unroll
      for (int r = 0; r < 8; ++r) {
        int row = m_blk + wm * 32 + ti * 16 + rb + r;
        int col = n_blk + wn * 32 + tj * 16 + cb;
        if (row < M && col < N) {
          float v = acc[ti][tj][r];
          if (bias) v += bias[col];
          if (act == 1) v = 0.5f * v * (1.0f + erff(v * 0.70710678118f));
          if (resid) v += resid[(long long)row * N + col];
          C[(long long)row * N + col] = v;
        }
      }
}

// ---------------------------------------------------------------
// Attention scores: S[b,h,qi,kj] = 0.125 * q . k   (WMMA, 32x32/block)
// ---------------------------------------------------------------
__global__ __launch_bounds__(128)
void attn_scores(const float* __restrict__ q, const float* __restrict__ k,
                 float* __restrict__ S, int Ntok, int Npad) {
  __shared__ __align__(32) unsigned short Qs[4 * 512];  // (rowsub, kslice)
  __shared__ __align__(32) unsigned short Ks[4 * 512];  // (keysub, kslice)
  const int tid  = threadIdx.x;
  const int lane = tid & 31;
  const int wid  = tid >> 5;
  const int bh = blockIdx.z, b = bh / HC, h = bh % HC;
  const int qbase = blockIdx.y * 32, kbase = blockIdx.x * 32;

  #pragma unroll
  for (int i = 0; i < 4; ++i) {
    int p = tid + i * 128;                 // 0..511 quads
    int r = p >> 4;                        // 0..31
    int cq = (p & 15) * 4;                 // 0,4,..,60
    int ks = cq >> 5, kl = cq & 31;
    // Q (A-fragments)
    int qi = qbase + r;
    int okq = qi < Ntok;
    const float4 vq = *(const float4*)(q + ((long long)b * Ntok + (okq ? qi : 0)) * DC + h * HDC + cq);
    unsigned int qx = pack2bf(vq.x, vq.y);
    unsigned int qy = pack2bf(vq.z, vq.w);
    uint2 pq;
    pq.x = okq ? qx : 0u;
    pq.y = okq ? qy : 0u;
    int qidx = ((r >> 4) * 2 + ks) * 512 + afrag_idx(r & 15, kl);
    *(uint2*)&Qs[qidx] = pq;
    // K (B-fragments: B[k][n] = K[key=n][dim=k])
    int ki = kbase + r;
    int okk = ki < Ntok;
    const float4 vk = *(const float4*)(k + ((long long)b * Ntok + (okk ? ki : 0)) * DC + h * HDC + cq);
    unsigned int kx = pack2bf(vk.x, vk.y);
    unsigned int ky = pack2bf(vk.z, vk.w);
    uint2 pkk;
    pkk.x = okk ? kx : 0u;
    pkk.y = okk ? ky : 0u;
    int kidx = ((r >> 4) * 2 + ks) * 512 + bfrag_idx(kl, r & 15);
    *(uint2*)&Ks[kidx] = pkk;
  }
  __syncthreads();

  const int wq = wid >> 1, wk = wid & 1;
  v8f c0 = {};
  #pragma unroll
  for (int ks2 = 0; ks2 < 2; ++ks2) {
    v16bf a  = ldfrag(Qs + (wq * 2 + ks2) * 512 + lane * 16);
    v16bf bb = ldfrag(Ks + (wk * 2 + ks2) * 512 + lane * 16);
    c0 = __builtin_amdgcn_wmma_f32_16x16x32_bf16(false, a, false, bb, (short)0, c0, false, false);
  }
  const int rb = (lane >> 4) * 8, cb = lane & 15;
  #pragma unroll
  for (int r = 0; r < 8; ++r) {
    int qi = qbase + wq * 16 + rb + r;
    int kj = kbase + wk * 16 + cb;
    S[(((long long)b * HC + h) * Npad + qi) * Npad + kj] = c0[r] * 0.125f;
  }
}

// ---------------------------------------------------------------
// Row softmax over Npad, masked at Ntok (one block per (b,h,qi))
// ---------------------------------------------------------------
__global__ __launch_bounds__(128)
void softmax_rows(float* __restrict__ S, int Ntok, int Npad) {
  const int row = blockIdx.x;              // over B*H*Ntok
  const int bh = row / Ntok, qi = row % Ntok;
  float* p = S + ((long long)bh * Npad + qi) * Npad;
  __shared__ float red[128];
  const int tid = threadIdx.x;

  float mx = -3.4e38f;
  for (int j = tid; j < Npad; j += 128)
    mx = fmaxf(mx, (j < Ntok) ? p[j] : -3.4e38f);
  red[tid] = mx; __syncthreads();
  for (int s = 64; s; s >>= 1) { if (tid < s) red[tid] = fmaxf(red[tid], red[tid + s]); __syncthreads(); }
  mx = red[0]; __syncthreads();

  float sum = 0.0f;
  for (int j = tid; j < Npad; j += 128) {
    float v = (j < Ntok) ? __expf(p[j] - mx) : 0.0f;
    p[j] = v; sum += v;
  }
  red[tid] = sum; __syncthreads();
  for (int s = 64; s; s >>= 1) { if (tid < s) red[tid] += red[tid + s]; __syncthreads(); }
  float inv = 1.0f / red[0]; __syncthreads();
  for (int j = tid; j < Npad; j += 128) p[j] *= inv;
}

// ---------------------------------------------------------------
// O = P @ V  (WMMA; block: 16 queries x 64 dims for one (b,h))
// ---------------------------------------------------------------
__global__ __launch_bounds__(128)
void attn_av(const float* __restrict__ P, const float* __restrict__ v,
             float* __restrict__ o, int Ntok, int Npad) {
  __shared__ __align__(32) unsigned short Ps[512];       // 16x32 A-fragment
  __shared__ __align__(32) unsigned short Vs[4 * 512];   // 4 d-subtiles of 32x16
  const int tid  = threadIdx.x;
  const int lane = tid & 31;
  const int wid  = tid >> 5;               // dim tile 0..3
  const int bh = blockIdx.y, b = bh / HC, h = bh % HC;
  const int qbase = blockIdx.x * 16;
  const float* prow = P + (long long)bh * Npad * Npad;

  v8f acc = {};
  for (int key0 = 0; key0 < Npad; key0 += 32) {
    // P tile 16x32: 128 quads (one per thread)
    {
      int r = tid >> 3;                    // 0..15
      int cq = (tid & 7) * 4;              // 0..28
      int qi = qbase + r;
      int ok = qi < Ntok;
      const float4 vp = *(const float4*)(prow + (long long)(ok ? qi : 0) * Npad + key0 + cq);
      unsigned int px = pack2bf(vp.x, vp.y);
      unsigned int py = pack2bf(vp.z, vp.w);
      uint2 pk;
      pk.x = ok ? px : 0u;
      pk.y = ok ? py : 0u;
      *(uint2*)&Ps[afrag_idx(r, cq)] = pk;
    }
    // V tile 32x64: quads along key axis
    #pragma unroll
    for (int i = 0; i < 4; ++i) {
      int p = tid + i * 128;               // 0..511
      int kq = (p >> 6) * 4;               // 0,4,..,28
      int d  = p & 63;
      float xv[4];
      #pragma unroll
      for (int j = 0; j < 4; ++j) {
        int ki = key0 + kq + j;
        int ok = ki < Ntok;
        float val = v[((long long)b * Ntok + (ok ? ki : 0)) * DC + h * HDC + d];
        xv[j] = ok ? val : 0.0f;           // cndmask on float
      }
      uint2 pk;
      pk.x = pack2bf(xv[0], xv[1]);
      pk.y = pack2bf(xv[2], xv[3]);
      int idx = (d >> 4) * 512 + bfrag_idx(kq, d & 15);
      *(uint2*)&Vs[idx] = pk;
    }
    __syncthreads();

    v16bf a  = ldfrag(Ps + lane * 16);
    v16bf bb = ldfrag(Vs + wid * 512 + lane * 16);
    acc = __builtin_amdgcn_wmma_f32_16x16x32_bf16(false, a, false, bb, (short)0, acc, false, false);
    __syncthreads();
  }
  const int rb = (lane >> 4) * 8;
  const int dc = wid * 16 + (lane & 15);
  #pragma unroll
  for (int r = 0; r < 8; ++r) {
    int qi = qbase + rb + r;
    if (qi < Ntok) o[((long long)b * Ntok + qi) * DC + h * HDC + dc] = acc[r];
  }
}

// ---------------------------------------------------------------
// LayerNorm over last dim (384); one wave per row, 8 rows/block
// ---------------------------------------------------------------
__global__ __launch_bounds__(256)
void layernorm384(const float* __restrict__ x, const float* __restrict__ g,
                  const float* __restrict__ bt, float* __restrict__ out, int M) {
  const int lane = threadIdx.x & 31;
  const int row = blockIdx.x * 8 + (threadIdx.x >> 5);
  if (row >= M) return;
  const float* p = x + (long long)row * DC;
  float vals[12];
  float s = 0.0f;
  #pragma unroll
  for (int i = 0; i < 12; ++i) { vals[i] = p[lane + i * 32]; s += vals[i]; }
  #pragma unroll
  for (int off = 16; off; off >>= 1) s += __shfl_xor(s, off);
  float mean = s * (1.0f / DC);
  float vs = 0.0f;
  #pragma unroll
  for (int i = 0; i < 12; ++i) { float d = vals[i] - mean; vs += d * d; }
  #pragma unroll
  for (int off = 16; off; off >>= 1) vs += __shfl_xor(vs, off);
  float inv = rsqrtf(vs * (1.0f / DC) + 1e-5f);
  float* q = out + (long long)row * DC;
  #pragma unroll
  for (int i = 0; i < 12; ++i) {
    int c = lane + i * 32;
    q[c] = (vals[i] - mean) * inv * g[c] + bt[c];
  }
}

// ---------------------------------------------------------------
// im2col for patch embed: (B,3,256,256) -> (B*256, 768)
// ---------------------------------------------------------------
__global__ void im2col_patch(const float* __restrict__ x, float* __restrict__ pc, int total) {
  int i = blockIdx.x * 256 + threadIdx.x;
  if (i >= total) return;
  int f = i % 768; int rest = i / 768;
  int patch = rest % 256; int b = rest / 256;
  int c = f / 256; int fin = f % 256; int py = fin / 16, px = fin % 16;
  int gy = patch / 16, gx = patch % 16;
  pc[i] = x[(((long long)b * 3 + c) * 256 + gy * 16 + py) * 256 + gx * 16 + px];
}

// cls + tok + pos_emb  -> t (B*257, 384)
__global__ void assemble_tokens(const float* __restrict__ tok, const float* __restrict__ cls,
                                const float* __restrict__ pos, float* __restrict__ t, int total) {
  int i = blockIdx.x * 256 + threadIdx.x;
  if (i >= total) return;
  int d = i % DC; int rest = i / DC;
  int n = rest % 257; int b = rest / 257;
  float v = (n == 0) ? cls[d] : tok[((long long)b * 256 + (n - 1)) * DC + d];
  t[i] = v + pos[n * DC + d];
}

// maxpool 3x3 stride 2 pad 1 on token grid, cls preserved
__global__ void down_tokens_k(const float* __restrict__ in, float* __restrict__ out,
                              int s, int total) {
  int i = blockIdx.x * 256 + threadIdx.x;
  if (i >= total) return;
  int so = (s - 1) / 2 + 1;
  int Nin = s * s + 1, Nout = so * so + 1;
  int d = i % DC; int rest = i / DC;
  int n = rest % Nout; int b = rest / Nout;
  float v;
  if (n == 0) v = in[(long long)b * Nin * DC + d];
  else {
    int op = n - 1, oy = op / so, ox = op % so;
    v = -3.4e38f;
    for (int ky = 0; ky < 3; ++ky) {
      int iy = oy * 2 - 1 + ky; if (iy < 0 || iy >= s) continue;
      for (int kx = 0; kx < 3; ++kx) {
        int ix = ox * 2 - 1 + kx; if (ix < 0 || ix >= s) continue;
        v = fmaxf(v, in[((long long)b * Nin + 1 + iy * s + ix) * DC + d]);
      }
    }
  }
  out[i] = v;
}

// bilinear 2x upsample on token grid, cls preserved
__global__ void up_tokens_k(const float* __restrict__ in, float* __restrict__ out,
                            int s, int total) {
  int i = blockIdx.x * 256 + threadIdx.x;
  if (i >= total) return;
  int so = 2 * s;
  int Nin = s * s + 1, Nout = so * so + 1;
  int d = i % DC; int rest = i / DC;
  int n = rest % Nout; int b = rest / Nout;
  float v;
  if (n == 0) v = in[(long long)b * Nin * DC + d];
  else {
    int op = n - 1, oy = op / so, ox = op % so;
    float scale = (float)(s - 1) / (float)(so - 1);
    float fy = oy * scale, fx = ox * scale;
    int y0 = (int)floorf(fy); int y1 = min(y0 + 1, s - 1); float wy = fy - (float)y0;
    int x0 = (int)floorf(fx); int x1 = min(x0 + 1, s - 1); float wx = fx - (float)x0;
    const float* base = in + ((long long)b * Nin + 1) * DC + d;
    float g00 = base[(long long)(y0 * s + x0) * DC], g01 = base[(long long)(y0 * s + x1) * DC];
    float g10 = base[(long long)(y1 * s + x0) * DC], g11 = base[(long long)(y1 * s + x1) * DC];
    v = (g00 * (1.0f - wx) + g01 * wx) * (1.0f - wy) + (g10 * (1.0f - wx) + g11 * wx) * wy;
  }
  out[i] = v;
}

// ---------------------------------------------------------------
extern "C" void kernel_launch(void* const* d_in, const int* in_sizes, int n_in,
                              void* d_out, int out_size, void* d_ws, size_t ws_size,
                              hipStream_t stream) {
  const float* xin    = (const float*)d_in[0];
  const float* conv_w = (const float*)d_in[1];
  const float* conv_b = (const float*)d_in[2];
  const float* clstk  = (const float*)d_in[3];
  const float* posemb = (const float*)d_in[4];
  const float* ln1_g  = (const float*)d_in[5];
  const float* ln1_b  = (const float*)d_in[6];
  const float* qw     = (const float*)d_in[7];
  const float* qbias  = (const float*)d_in[8];
  const float* kw     = (const float*)d_in[9];
  const float* kbias  = (const float*)d_in[10];
  const float* vw     = (const float*)d_in[11];
  const float* vbias  = (const float*)d_in[12];
  const float* pwm    = (const float*)d_in[13];
  const float* pbias  = (const float*)d_in[14];
  const float* ln2_g  = (const float*)d_in[15];
  const float* ln2_b  = (const float*)d_in[16];
  const float* fc1w   = (const float*)d_in[17];
  const float* fc1b   = (const float*)d_in[18];
  const float* fc2w   = (const float*)d_in[19];
  const float* fc2b   = (const float*)d_in[20];
  const float* normg  = (const float*)d_in[21];
  const float* normb  = (const float*)d_in[22];

  // ---- workspace carve ----
  char* wsp = (char*)d_ws;
  auto alloc = [&](size_t nf) { float* p = (float*)wsp; wsp += nf * sizeof(float); return p; };
  const size_t CAP = (size_t)BC * 288 * DC;        // max tokens padded
  float* imcol = alloc((size_t)BC * 256 * 768);
  float* xA    = alloc(CAP);
  float* xB    = alloc(CAP);
  float* xnb   = alloc(CAP);
  float* ynb   = alloc(CAP);
  float* qbuf  = alloc(CAP);
  float* kbuf  = alloc(CAP);
  float* vbuf  = alloc(CAP);
  float* obuf  = alloc(CAP);
  float* Sbuf  = alloc((size_t)BC * HC * 288 * 288);
  float* hbuf  = alloc((size_t)BC * 288 * HIDC);
  float* sv0   = alloc(CAP);
  float* sv1   = alloc(CAP);
  float* sv2   = alloc(CAP);
  float* sv3   = alloc(CAP);

  auto gemm = [&](const float* A, const float* Wm, const float* bias, const float* resid,
                  float* C, int M, int N, int K, int wt, int act) {
    dim3 g((N + 63) / 64, (M + 127) / 128);
    gemm_bf16_wmma<<<g, 256, 0, stream>>>(A, Wm, bias, resid, C, M, N, K, wt, act);
  };
  auto ln = [&](const float* in, const float* g, const float* b, float* out, int M) {
    layernorm384<<<(M + 7) / 8, 256, 0, stream>>>(in, g, b, out, M);
  };

  auto run_block = [&](int blk, float* x, const float* y, int Ntok) {
    const int M = BC * Ntok;
    const int Npad = (Ntok + 31) / 32 * 32;
    ln(x, ln1_g + blk * DC, ln1_b + blk * DC, xnb, M);
    const float* ykv = xnb;
    if (y) { ln(y, ln1_g + blk * DC, ln1_b + blk * DC, ynb, M); ykv = ynb; }
    gemm(xnb, qw + (size_t)blk * DC * DC, qbias + blk * DC, nullptr, qbuf, M, DC, DC, 0, 0);
    gemm(ykv, kw + (size_t)blk * DC * DC, kbias + blk * DC, nullptr, kbuf, M, DC, DC, 0, 0);
    gemm(ykv, vw + (size_t)blk * DC * DC, vbias + blk * DC, nullptr, vbuf, M, DC, DC, 0, 0);
    dim3 gs(Npad / 32, Npad / 32, BC * HC);
    attn_scores<<<gs, 128, 0, stream>>>(qbuf, kbuf, Sbuf, Ntok, Npad);
    softmax_rows<<<BC * HC * Ntok, 128, 0, stream>>>(Sbuf, Ntok, Npad);
    dim3 ga((Ntok + 15) / 16, BC * HC);
    attn_av<<<ga, 128, 0, stream>>>(Sbuf, vbuf, obuf, Ntok, Npad);
    gemm(obuf, pwm + (size_t)blk * DC * DC, pbias + blk * DC, x, x, M, DC, DC, 0, 0);
    ln(x, ln2_g + blk * DC, ln2_b + blk * DC, xnb, M);
    gemm(xnb, fc1w + (size_t)blk * DC * HIDC, fc1b + blk * HIDC, nullptr, hbuf, M, HIDC, DC, 0, 1);
    gemm(hbuf, fc2w + (size_t)blk * HIDC * DC, fc2b + blk * DC, x, x, M, DC, HIDC, 0, 0);
  };

  auto save = [&](float* dst, const float* src, int Ntok) {
    hipMemcpyAsync(dst, src, (size_t)BC * Ntok * DC * sizeof(float),
                   hipMemcpyDeviceToDevice, stream);
  };
  auto down = [&](float*& xcur, float*& xalt, int s) {
    int so = (s - 1) / 2 + 1; int Nout = so * so + 1;
    int tot = BC * Nout * DC;
    down_tokens_k<<<(tot + 255) / 256, 256, 0, stream>>>(xcur, xalt, s, tot);
    float* t = xcur; xcur = xalt; xalt = t;
    return Nout;
  };
  auto up = [&](float*& xcur, float*& xalt, int s) {
    int so = 2 * s; int Nout = so * so + 1;
    int tot = BC * Nout * DC;
    up_tokens_k<<<(tot + 255) / 256, 256, 0, stream>>>(xcur, xalt, s, tot);
    float* t = xcur; xcur = xalt; xalt = t;
    return Nout;
  };

  // ---- patch embed + tokens ----
  int tot = BC * 256 * 768;
  im2col_patch<<<(tot + 255) / 256, 256, 0, stream>>>(xin, imcol, tot);
  gemm(imcol, conv_w, conv_b, nullptr, qbuf, BC * 256, DC, 768, /*wt=*/1, 0);
  tot = BC * 257 * DC;
  float* xcur = xA; float* xalt = xB;
  assemble_tokens<<<(tot + 255) / 256, 256, 0, stream>>>(qbuf, clstk, posemb, xcur, tot);

  // ---- Stage 1 (down), blocks 0-5 ----
  run_block(0, xcur, nullptr, 257);
  run_block(1, xcur, nullptr, 257);
  save(sv0, xcur, 257);
  int N = down(xcur, xalt, 16);            // 257 -> 65
  run_block(2, xcur, nullptr, N);
  run_block(3, xcur, nullptr, N);
  save(sv1, xcur, N);
  N = down(xcur, xalt, 8);                 // 65 -> 17
  run_block(4, xcur, nullptr, N);
  run_block(5, xcur, nullptr, N);

  // ---- Stage 2 (up), blocks 6-11, prev=[sv0@257, sv1@65] ----
  run_block(6, xcur, nullptr, 17);
  run_block(7, xcur, nullptr, 17);
  save(sv2, xcur, 17);
  N = up(xcur, xalt, 4);                   // 17 -> 65
  run_block(8, xcur, sv1, N);
  run_block(9, xcur, nullptr, N);
  save(sv3, xcur, N);
  N = up(xcur, xalt, 8);                   // 65 -> 257
  run_block(10, xcur, sv0, N);
  run_block(11, xcur, nullptr, N);

  // ---- Stage 3 (down), blocks 12-17, prev=[sv2@17, sv3@65] ----
  run_block(12, xcur, nullptr, 257);
  run_block(13, xcur, nullptr, 257);
  N = down(xcur, xalt, 16);                // 257 -> 65
  run_block(14, xcur, sv3, N);
  run_block(15, xcur, nullptr, N);
  N = down(xcur, xalt, 8);                 // 65 -> 17
  run_block(16, xcur, sv2, N);
  run_block(17, xcur, nullptr, N);

  // ---- final layernorm into d_out ----
  ln(xcur, normg, normb, (float*)d_out, BC * 17);

  (void)in_sizes; (void)n_in; (void)out_size; (void)ws_size;
}
